// Transformer_25829933318658
// MI455X (gfx1250) — matrix operand order repaired and checked by reading, not compiled, not run on hardware
//
#include <hip/hip_runtime.h>
#include <hip/hip_bf16.h>

// ---------------------------------------------------------------------------
// Types for CDNA5 WMMA (gfx1250, wave32)
// ---------------------------------------------------------------------------
typedef __attribute__((ext_vector_type(16))) __bf16 v16bf;
typedef __attribute__((ext_vector_type(8)))  float  v8f;

// Async global->LDS builtin takes pointers to generic int4 vectors in
// AS(1)/AS(3) (diagnostic-confirmed: param pointee is vector_size(16) int).
typedef int v4i_gcc __attribute__((vector_size(16)));
typedef __attribute__((address_space(1))) v4i_gcc gv4i_t;
typedef __attribute__((address_space(3))) v4i_gcc lv4i_t;

// CDNA5 async global->LDS DMA path (ASYNCcnt), if the toolchain exposes it.
#if defined(__has_builtin)
#if __has_builtin(__builtin_amdgcn_global_load_async_to_lds_b128) && \
    __has_builtin(__builtin_amdgcn_s_wait_asynccnt)
#define HAVE_ASYNC_LDS 1
#endif
#endif
#ifndef HAVE_ASYNC_LDS
#define HAVE_ASYNC_LDS 0
#endif

union Frag {
    v16bf bf;
    uint4 q[2];
};

__device__ __forceinline__ v8f vzero8() {
    v8f z;
#pragma unroll
    for (int i = 0; i < 8; ++i) z[i] = 0.0f;
    return z;
}

__device__ __forceinline__ unsigned short f2bf(float f) {
    unsigned int u = __float_as_uint(f);
    if ((u & 0x7F800000u) == 0x7F800000u) {          // Inf/NaN: quiet, no round
        return (unsigned short)((u | 0x00400000u) >> 16);
    }
    unsigned int r = u + 0x7FFFu + ((u >> 16) & 1u); // round-to-nearest-even
    return (unsigned short)(r >> 16);
}

__device__ __forceinline__ float bf2f(unsigned short h) {
    return __uint_as_float(((unsigned int)h) << 16);
}

// Model constants
#define BB 2
#define SS 1024
#define DD 1024
#define HH 8
#define HD 128
#define DEPTH 8
#define FFD 4096
#define BS (BB * SS)

// ---------------------------------------------------------------------------
// Weight convert + transpose: W (K x N, f32) -> Wt (N x K, bf16), per layer z
// ---------------------------------------------------------------------------
__global__ __launch_bounds__(256) void wtrans_kernel(
    const float* __restrict__ W, unsigned short* __restrict__ Wt, int K, int N)
{
    __shared__ float tile[32][33];
    const int z = blockIdx.z;
    const float* Wz = W + (size_t)z * K * N;
    unsigned short* Wtz = Wt + (size_t)z * N * K;
    const int n0 = blockIdx.x * 32, k0 = blockIdx.y * 32;
    const int tx = threadIdx.x, ty = threadIdx.y;      // block (32, 8)
#pragma unroll
    for (int i = 0; i < 4; ++i)
        tile[ty + i * 8][tx] = Wz[(size_t)(k0 + ty + i * 8) * N + n0 + tx];
    __syncthreads();
#pragma unroll
    for (int i = 0; i < 4; ++i)
        Wtz[(size_t)(n0 + ty + i * 8) * K + k0 + tx] = f2bf(tile[tx][ty + i * 8]);
}

// ---------------------------------------------------------------------------
// LayerNorm over D=1024, one block (256 thr) per row.
// Optionally writes the pre-LN row (context) into new_state[(row*8+layer)*D].
// ---------------------------------------------------------------------------
__global__ __launch_bounds__(256) void ln_bf16_kernel(
    const float* __restrict__ x, const float* __restrict__ g,
    const float* __restrict__ bp, unsigned short* __restrict__ out,
    float* __restrict__ state, int layer)
{
    __shared__ float red[256];
    const int r = blockIdx.x, tid = threadIdx.x;
    const size_t base = (size_t)r * DD;
    float v4[4];
    float s = 0.f;
#pragma unroll
    for (int i = 0; i < 4; ++i) { v4[i] = x[base + tid + i * 256]; s += v4[i]; }
    red[tid] = s; __syncthreads();
    for (int o = 128; o > 0; o >>= 1) { if (tid < o) red[tid] += red[tid + o]; __syncthreads(); }
    const float mean = red[0] * (1.0f / DD);
    __syncthreads();
    float vs = 0.f;
#pragma unroll
    for (int i = 0; i < 4; ++i) { float d = v4[i] - mean; vs += d * d; }
    red[tid] = vs; __syncthreads();
    for (int o = 128; o > 0; o >>= 1) { if (tid < o) red[tid] += red[tid + o]; __syncthreads(); }
    const float rstd = rsqrtf(red[0] * (1.0f / DD) + 1e-5f);
#pragma unroll
    for (int i = 0; i < 4; ++i) {
        const int d = tid + i * 256;
        out[base + d] = f2bf((v4[i] - mean) * rstd * g[d] + bp[d]);
        if (state) state[((size_t)r * DEPTH + layer) * DD + d] = v4[i];
    }
}

__global__ __launch_bounds__(256) void ln_f32_kernel(
    const float* __restrict__ x, const float* __restrict__ g,
    const float* __restrict__ bp, float* __restrict__ out)
{
    __shared__ float red[256];
    const int r = blockIdx.x, tid = threadIdx.x;
    const size_t base = (size_t)r * DD;
    float v4[4];
    float s = 0.f;
#pragma unroll
    for (int i = 0; i < 4; ++i) { v4[i] = x[base + tid + i * 256]; s += v4[i]; }
    red[tid] = s; __syncthreads();
    for (int o = 128; o > 0; o >>= 1) { if (tid < o) red[tid] += red[tid + o]; __syncthreads(); }
    const float mean = red[0] * (1.0f / DD);
    __syncthreads();
    float vs = 0.f;
#pragma unroll
    for (int i = 0; i < 4; ++i) { float d = v4[i] - mean; vs += d * d; }
    red[tid] = vs; __syncthreads();
    for (int o = 128; o > 0; o >>= 1) { if (tid < o) red[tid] += red[tid + o]; __syncthreads(); }
    const float rstd = rsqrtf(red[0] * (1.0f / DD) + 1e-5f);
#pragma unroll
    for (int i = 0; i < 4; ++i) {
        const int d = tid + i * 256;
        out[base + d] = (v4[i] - mean) * rstd * g[d] + bp[d];
    }
}

__global__ __launch_bounds__(256) void zero_kernel(float* __restrict__ p, int n) {
    for (int i = threadIdx.x + blockIdx.x * 256; i < n; i += gridDim.x * 256) p[i] = 0.f;
}

// ---------------------------------------------------------------------------
// bf16 WMMA GEMM: Out(MxN) = A(MxK,bf16) @ Bt(NxK,bf16)^T  (+bias)(+resid)(gelu)
// Block tile 128x128, BK=64, 256 threads = 8 waves, each wave 32x64 (8 tiles).
// Data movement: async global->LDS DMA with double-buffered tiles (ASYNCcnt)
// when available; otherwise a register-prefetch software pipeline.
// Dynamic LDS: 4 * 128*72 bf16 = 73728 B (host always passes this size).
// ---------------------------------------------------------------------------
#define GEMM_LDS_BYTES (4 * 128 * 72 * 2)

template<bool OUTBF16, bool BIAS, bool RESID, bool GELU>
__global__ __launch_bounds__(256) void gemm_wmma_kernel(
    const unsigned short* __restrict__ A,   // M x K
    const unsigned short* __restrict__ Bt,  // N x K (pre-transposed weights)
    void* __restrict__ Out,                 // M x N
    const float* __restrict__ bias,
    const float* __restrict__ resid,
    int K, int N, float scale)
{
    extern __shared__ __align__(16) char gsm[];
    unsigned short* As0 = (unsigned short*)gsm;      // 128*72
    unsigned short* Bs0 = As0 + 128 * 72;
    unsigned short* As1 = Bs0 + 128 * 72;            // async double-buffer
    unsigned short* Bs1 = As1 + 128 * 72;

    const int tid = threadIdx.x;
    const int w = tid >> 5, lane = tid & 31, lo = lane & 15, hi = lane >> 4;
    const int m0 = blockIdx.y * 128, n0 = blockIdx.x * 128;
    const int mbase = (w & 3) * 32, nbase = (w >> 2) * 64;
    const int r0 = tid >> 3, ch = tid & 7;           // tile-copy assignment
    const int nk = K >> 6;

    v8f acc[2][4];
#pragma unroll
    for (int mt = 0; mt < 2; ++mt)
#pragma unroll
        for (int nt = 0; nt < 4; ++nt) acc[mt][nt] = vzero8();

    auto compute = [&](const unsigned short* Asp, const unsigned short* Bsp) {
#pragma unroll
        for (int kk = 0; kk < 64; kk += 32) {
            Frag a[2], bf[4];
#pragma unroll
            for (int mt = 0; mt < 2; ++mt) {
                const int ab = (mbase + mt * 16 + lo) * 72 + kk;
                a[mt].q[0] = *(const uint4*)&Asp[ab + hi * 16];
                a[mt].q[1] = *(const uint4*)&Asp[ab + hi * 16 + 8];
            }
#pragma unroll
            for (int nt = 0; nt < 4; ++nt) {
                const int bb = (nbase + nt * 16 + lo) * 72 + kk + hi * 16;
                bf[nt].q[0] = *(const uint4*)&Bsp[bb];
                bf[nt].q[1] = *(const uint4*)&Bsp[bb + 8];
            }
#pragma unroll
            for (int mt = 0; mt < 2; ++mt)
#pragma unroll
                for (int nt = 0; nt < 4; ++nt)
                    acc[mt][nt] = __builtin_amdgcn_wmma_f32_16x16x32_bf16(
                        false, a[mt].bf, false, bf[nt].bf,
                        (short)0, acc[mt][nt], false, false);
        }
    };

#if HAVE_ASYNC_LDS
    // ---- Async DMA path: double-buffered LDS tiles, ASYNCcnt tracked ----
    auto issue = [&](int k0g, unsigned short* Asp, unsigned short* Bsp) {
#pragma unroll
        for (int i = 0; i < 4; ++i) {
            const int row = r0 + 32 * i;
            __builtin_amdgcn_global_load_async_to_lds_b128(
                (gv4i_t*)(A + (size_t)(m0 + row) * K + k0g + ch * 8),
                (lv4i_t*)&Asp[row * 72 + ch * 8], 0, 0);
            __builtin_amdgcn_global_load_async_to_lds_b128(
                (gv4i_t*)(Bt + (size_t)(n0 + row) * K + k0g + ch * 8),
                (lv4i_t*)&Bsp[row * 72 + ch * 8], 0, 0);
        }
    };
    issue(0, As0, Bs0);
    for (int t = 0; t < nk; ++t) {
        const int buf = t & 1;
        unsigned short* Asp = buf ? As1 : As0;
        unsigned short* Bsp = buf ? Bs1 : Bs0;
        if (t + 1 < nk) {
            issue((t + 1) << 6, buf ? As0 : As1, buf ? Bs0 : Bs1);
            // 8 loads for tile t+1 outstanding; in-order completion means
            // waiting to <=8 guarantees tile t has fully landed in LDS.
            __builtin_amdgcn_s_wait_asynccnt(8);
        } else {
            __builtin_amdgcn_s_wait_asynccnt(0);
        }
        __syncthreads();
        compute(Asp, Bsp);
        __syncthreads();
    }
#else
    // ---- Fallback: register-prefetch software pipeline, single buffer ----
    uint4 ra[4], rb[4];
    auto ldg = [&](int k0g) {
#pragma unroll
        for (int i = 0; i < 4; ++i) {
            const int row = r0 + 32 * i;
            ra[i] = *(const uint4*)&A[(size_t)(m0 + row) * K + k0g + ch * 8];
            rb[i] = *(const uint4*)&Bt[(size_t)(n0 + row) * K + k0g + ch * 8];
        }
    };
    ldg(0);
    for (int t = 0; t < nk; ++t) {
        __syncthreads();
#pragma unroll
        for (int i = 0; i < 4; ++i) {
            const int row = r0 + 32 * i;
            *(uint4*)&As0[row * 72 + ch * 8] = ra[i];
            *(uint4*)&Bs0[row * 72 + ch * 8] = rb[i];
        }
        __syncthreads();
        if (t + 1 < nk) ldg((t + 1) << 6);   // overlap next global tile
        compute(As0, Bs0);
    }
#endif

    // Epilogue
#pragma unroll
    for (int mt = 0; mt < 2; ++mt) {
#pragma unroll
        for (int nt = 0; nt < 4; ++nt) {
            const int col = n0 + nbase + nt * 16 + lo;
            const float bv = BIAS ? bias[col] : 0.0f;
#pragma unroll
            for (int r = 0; r < 8; ++r) {
                const int grow = m0 + mbase + mt * 16 + r + hi * 8;
                float v = acc[mt][nt][r] * scale + bv;
                if (RESID) v += resid[(size_t)grow * N + col];
                if (GELU)  v = 0.5f * v * (1.0f + erff(v * 0.70710678118654752f));
                if (OUTBF16)
                    ((unsigned short*)Out)[(size_t)grow * N + col] = f2bf(v);
                else
                    ((float*)Out)[(size_t)grow * N + col] = v;
            }
        }
    }
}

// ---------------------------------------------------------------------------
// Causal attention, one block per (b, h, 32-row q tile). All scores kept in
// LDS (CDNA5: 320KB/WGP), WMMA for QK^T and PV, fused softmax, and per-column
// normalized-prob sums atomically added into attention_maps.
// ---------------------------------------------------------------------------
#define SSTR 1032
#define QSTR 136
#define KSTR 136

__global__ __launch_bounds__(256) void attn_kernel(
    const unsigned short* __restrict__ q,   // (BS, D) bf16, pre-scaled by HD^-0.5
    const unsigned short* __restrict__ kv,  // (BS, 2D) bf16: [k | v]
    unsigned short* __restrict__ y,         // (BS, D) bf16
    float* __restrict__ amap)               // (B*S) f32
{
    extern __shared__ __align__(16) char smem[];
    float* sc = (float*)smem;                                     // 32 x SSTR f32
    unsigned short* P  = (unsigned short*)(smem + 32 * SSTR * 4); // 32 x SSTR bf16
    unsigned short* qt = P + 32 * SSTR;                           // 32 x QSTR
    unsigned short* kt = qt + 32 * QSTR;                          // 128 x KSTR
    float* inv = (float*)(kt + 128 * KSTR);                       // 32 f32

    const int tid = threadIdx.x;
    const int w = tid >> 5, lane = tid & 31, lo = lane & 15, hi = lane >> 4;
    const int qti = blockIdx.x, h = blockIdx.y, b = blockIdx.z;
    const int q0 = qti * 32;
    const int nkb = (q0 + 32 + 127) >> 7;
    const size_t rowbase = (size_t)b * SS;
    const int mt = w & 1, nb = (w >> 1) * 32;

    // Load Q tile (32 x 128)
    for (int c = tid; c < 512; c += 256) {
        const int r = c >> 4, ch = c & 15;
        *(uint4*)&qt[r * QSTR + ch * 8] =
            *(const uint4*)&q[(rowbase + q0 + r) * DD + h * HD + ch * 8];
    }

    // ---- Scores: S = Q @ K^T over all needed k blocks ----
    for (int kb = 0; kb < nkb; ++kb) {
        __syncthreads();
        for (int c = tid; c < 2048; c += 256) {
            const int r = c >> 4, ch = c & 15;
            *(uint4*)&kt[r * KSTR + ch * 8] =
                *(const uint4*)&kv[(rowbase + kb * 128 + r) * (2 * DD) + h * HD + ch * 8];
        }
        __syncthreads();
        v8f acc[2] = { vzero8(), vzero8() };
#pragma unroll
        for (int kk = 0; kk < 128; kk += 32) {
            Frag a;
            const int ab = (mt * 16 + lo) * QSTR + kk;
            a.q[0] = *(const uint4*)&qt[ab + hi * 16];
            a.q[1] = *(const uint4*)&qt[ab + hi * 16 + 8];
#pragma unroll
            for (int nt = 0; nt < 2; ++nt) {
                Frag bf;
                const int bb = (nb + nt * 16 + lo) * KSTR + kk + hi * 16;
                bf.q[0] = *(const uint4*)&kt[bb];
                bf.q[1] = *(const uint4*)&kt[bb + 8];
                acc[nt] = __builtin_amdgcn_wmma_f32_16x16x32_bf16(
                    false, a.bf, false, bf.bf, (short)0, acc[nt], false, false);
            }
        }
#pragma unroll
        for (int nt = 0; nt < 2; ++nt)
#pragma unroll
            for (int r = 0; r < 8; ++r)
                sc[(mt * 16 + r + hi * 8) * SSTR + kb * 128 + nb + nt * 16 + lo] = acc[nt][r];
    }
    __syncthreads();

    // ---- Softmax (causal). 8 threads per row. ----
    const int row = tid >> 3, sub = tid & 7;
    const int kmax = q0 + row;
    const int ncols = nkb * 128;
    float m = -3.4e38f;
    for (int k = sub; k <= kmax; k += 8) m = fmaxf(m, sc[row * SSTR + k]);
    m = fmaxf(m, __shfl_xor(m, 1, 8));
    m = fmaxf(m, __shfl_xor(m, 2, 8));
    m = fmaxf(m, __shfl_xor(m, 4, 8));
    float s = 0.f;
    for (int k = sub; k < ncols; k += 8) {
        const float e = (k <= kmax) ? __expf(sc[row * SSTR + k] - m) : 0.0f;
        P[row * SSTR + k] = f2bf(e);
        s += e;
    }
    s += __shfl_xor(s, 1, 8);
    s += __shfl_xor(s, 2, 8);
    s += __shfl_xor(s, 4, 8);
    if (sub == 0) inv[row] = 1.0f / s;
    __syncthreads();

    // ---- attention_maps: column sums of normalized probs ----
    for (int col = tid; col < ncols; col += 256) {
        float a = 0.f;
#pragma unroll
        for (int r2 = 0; r2 < 32; ++r2) a += bf2f(P[r2 * SSTR + col]) * inv[r2];
        atomicAdd(&amap[b * SS + col], a);
    }

    // ---- O = P @ V (V loaded transposed into LDS: kt[hd][kpos]) ----
    v8f oacc[2] = { vzero8(), vzero8() };
    for (int kb = 0; kb < nkb; ++kb) {
        __syncthreads();
        for (int c = tid; c < 2048; c += 256) {
            const int r = c >> 4, ch = c & 15;
            uint4 dv = *(const uint4*)&kv[(rowbase + kb * 128 + r) * (2 * DD)
                                          + DD + h * HD + ch * 8];
            const unsigned short* pv = (const unsigned short*)&dv;
#pragma unroll
            for (int i2 = 0; i2 < 8; ++i2) kt[(ch * 8 + i2) * KSTR + r] = pv[i2];
        }
        __syncthreads();
#pragma unroll
        for (int kk = 0; kk < 128; kk += 32) {
            Frag a;
            const int ab = (mt * 16 + lo) * SSTR + kb * 128 + kk;
            a.q[0] = *(const uint4*)&P[ab + hi * 16];
            a.q[1] = *(const uint4*)&P[ab + hi * 16 + 8];
#pragma unroll
            for (int nt = 0; nt < 2; ++nt) {
                Frag bf;
                const int bb = (nb + nt * 16 + lo) * KSTR + kk + hi * 16;
                bf.q[0] = *(const uint4*)&kt[bb];
                bf.q[1] = *(const uint4*)&kt[bb + 8];
                oacc[nt] = __builtin_amdgcn_wmma_f32_16x16x32_bf16(
                    false, a.bf, false, bf.bf, (short)0, oacc[nt], false, false);
            }
        }
    }
#pragma unroll
    for (int nt = 0; nt < 2; ++nt)
#pragma unroll
        for (int r = 0; r < 8; ++r) {
            const int rl = mt * 16 + r + hi * 8;
            const float v = oacc[nt][r] * inv[rl];
            y[(rowbase + q0 + rl) * DD + h * HD + nb + nt * 16 + lo] = f2bf(v);
        }
}

// ---------------------------------------------------------------------------
// Host orchestration
// ---------------------------------------------------------------------------
extern "C" void kernel_launch(void* const* d_in, const int* in_sizes, int n_in,
                              void* d_out, int out_size, void* d_ws, size_t ws_size,
                              hipStream_t stream) {
    const float* token = (const float*)d_in[0];
    const float* Wq    = (const float*)d_in[1];
    const float* Wkv   = (const float*)d_in[2];
    const float* Wo    = (const float*)d_in[3];
    const float* b_o   = (const float*)d_in[4];
    const float* g1    = (const float*)d_in[5];
    const float* b1    = (const float*)d_in[6];
    const float* g3    = (const float*)d_in[7];
    const float* b3    = (const float*)d_in[8];
    const float* W1    = (const float*)d_in[9];
    const float* W2    = (const float*)d_in[10];
    const float* g_out = (const float*)d_in[11];
    const float* b_out = (const float*)d_in[12];

    // Output regions: x_out (B,S,D) | new_state (B,S,DEPTH,D) | amap (B,S)
    float* out_x     = (float*)d_out;
    float* out_state = out_x + (size_t)BS * DD;
    float* out_amap  = out_state + (size_t)BS * DEPTH * DD;

    // Workspace layout
    char* p = (char*)d_ws;
    unsigned short* wq_t  = (unsigned short*)p; p += (size_t)DEPTH * DD * DD * 2;
    unsigned short* wkv_t = (unsigned short*)p; p += (size_t)DEPTH * 2 * DD * DD * 2;
    unsigned short* wo_t  = (unsigned short*)p; p += (size_t)DEPTH * DD * DD * 2;
    unsigned short* w1_t  = (unsigned short*)p; p += (size_t)DEPTH * FFD * DD * 2;
    unsigned short* w2_t  = (unsigned short*)p; p += (size_t)DEPTH * DD * FFD * 2;
    float* x              = (float*)p;          p += (size_t)BS * DD * 4;
    unsigned short* xn    = (unsigned short*)p; p += (size_t)BS * DD * 2;
    unsigned short* qb    = (unsigned short*)p; p += (size_t)BS * DD * 2;
    unsigned short* kvb   = (unsigned short*)p; p += (size_t)BS * 2 * DD * 2;
    unsigned short* yb    = (unsigned short*)p; p += (size_t)BS * DD * 2;
    unsigned short* h1    = (unsigned short*)p; p += (size_t)BS * FFD * 2;

    // x = token
    (void)hipMemcpyAsync(x, token, (size_t)BS * DD * sizeof(float),
                         hipMemcpyDeviceToDevice, stream);

    // Convert+transpose all weights fp32 -> bf16 (whole bf16 set ~192MB = L2)
    dim3 tb(32, 8);
    wtrans_kernel<<<dim3(DD / 32,      DD / 32,  DEPTH), tb, 0, stream>>>(Wq,  wq_t,  DD,  DD);
    wtrans_kernel<<<dim3(2 * DD / 32,  DD / 32,  DEPTH), tb, 0, stream>>>(Wkv, wkv_t, DD,  2 * DD);
    wtrans_kernel<<<dim3(DD / 32,      DD / 32,  DEPTH), tb, 0, stream>>>(Wo,  wo_t,  DD,  DD);
    wtrans_kernel<<<dim3(FFD / 32,     DD / 32,  DEPTH), tb, 0, stream>>>(W1,  w1_t,  DD,  FFD);
    wtrans_kernel<<<dim3(DD / 32,      FFD / 32, DEPTH), tb, 0, stream>>>(W2,  w2_t,  FFD, DD);

    zero_kernel<<<4, 256, 0, stream>>>(out_amap, BS);

    const float qscale = 0.08838834764831845f;   // HD^-0.5
    const size_t asmem = (size_t)32 * SSTR * 4 + 32 * SSTR * 2
                       + 32 * QSTR * 2 + 128 * KSTR * 2 + 32 * 4;
    const size_t gsmem = GEMM_LDS_BYTES;

    for (int l = 0; l < DEPTH; ++l) {
        // LN1 (+ context copy into new_state)
        ln_bf16_kernel<<<BS, 256, 0, stream>>>(x, g1 + l * DD, b1 + l * DD, xn,
                                               out_state, l);
        // q = (LN1 @ Wq) * scale ; kv = LN1 @ Wkv
        gemm_wmma_kernel<true, false, false, false>
            <<<dim3(DD / 128, BS / 128), 256, gsmem, stream>>>(
                xn, wq_t + (size_t)l * DD * DD, qb, nullptr, nullptr, DD, DD, qscale);
        gemm_wmma_kernel<true, false, false, false>
            <<<dim3(2 * DD / 128, BS / 128), 256, gsmem, stream>>>(
                xn, wkv_t + (size_t)l * 2 * DD * DD, kvb, nullptr, nullptr,
                DD, 2 * DD, 1.0f);
        // Attention (+ attention_maps accumulation)
        attn_kernel<<<dim3(SS / 32, HH, BB), 256, asmem, stream>>>(qb, kvb, yb,
                                                                   out_amap);
        // x = x + y @ Wo + b_o
        gemm_wmma_kernel<false, true, true, false>
            <<<dim3(DD / 128, BS / 128), 256, gsmem, stream>>>(
                yb, wo_t + (size_t)l * DD * DD, x, b_o + l * DD, x, DD, DD, 1.0f);
        // LN2
        ln_bf16_kernel<<<BS, 256, 0, stream>>>(x, g3 + l * DD, b3 + l * DD, xn,
                                               nullptr, 0);
        // h1 = gelu(LN2 @ W1)
        gemm_wmma_kernel<true, false, false, true>
            <<<dim3(FFD / 128, BS / 128), 256, gsmem, stream>>>(
                xn, w1_t + (size_t)l * FFD * DD, h1, nullptr, nullptr,
                DD, FFD, 1.0f);
        // x = x + h1 @ W2
        gemm_wmma_kernel<false, false, true, false>
            <<<dim3(DD / 128, BS / 128), 256, gsmem, stream>>>(
                h1, w2_t + (size_t)l * DD * FFD, x, nullptr, x, FFD, DD, 1.0f);
    }

    // Final LN -> x_out
    ln_f32_kernel<<<BS, 256, 0, stream>>>(x, g_out, b_out, out_x);
}
